// SelfDepAttention_75514114998865
// MI455X (gfx1250) — matrix operand order repaired and checked by reading, not compiled, not run on hardware
//
#include <hip/hip_runtime.h>

typedef _Float16 v16h __attribute__((ext_vector_type(16)));
typedef _Float16 v8h  __attribute__((ext_vector_type(8)));
typedef float    v8f  __attribute__((ext_vector_type(8)));
typedef float    v4f  __attribute__((ext_vector_type(4)));

#define Bc 2
#define Nc 512
#define Fc 64
#define HDc 256
#define ASTRIDE 264   // 264 halves per row: 132 dwords -> rows land on rotated banks

// ---------------------------------------------------------------------------
// Kernel 1: q = x@Wq + bq, k = x@Wk + bk  (v is dead code in the reference)
// Stores q (f32), k (f32) and k (f16) into workspace.
// ---------------------------------------------------------------------------
__global__ __launch_bounds__(256) void proj_kernel(
    const float* __restrict__ x,
    const float* __restrict__ Wq_w, const float* __restrict__ Wq_b,
    const float* __restrict__ Wk_w, const float* __restrict__ Wk_b,
    float* __restrict__ qf, float* __restrict__ kf, _Float16* __restrict__ kh) {
  __shared__ float xs[Fc];
  const int row = blockIdx.x;        // b*N + n
  const int d   = threadIdx.x;       // 0..255 = output feature
  if (d < Fc) xs[d] = x[row * Fc + d];
  __syncthreads();
  float aq = Wq_b[d];
  float ak = Wk_b[d];
#pragma unroll 8
  for (int f = 0; f < Fc; ++f) {
    const float xv = xs[f];
    aq = fmaf(xv, Wq_w[f * HDc + d], aq);
    ak = fmaf(xv, Wk_w[f * HDc + d], ak);
  }
  qf[row * HDc + d] = aq;
  kf[row * HDc + d] = ak;
  kh[row * HDc + d] = (_Float16)ak;
}

// ---------------------------------------------------------------------------
// A-fragment loader: 16-bit A matrix 16x32 layout (CDNA5 ISA 7.12.2):
//   lane half h: elements 0..7  -> K = 8h + e
//                elements 8..15 -> K = 16 + 8h + e
// Two 16B ds_load_b128 per fragment.
// ---------------------------------------------------------------------------
__device__ __forceinline__ v16h load_afrag(const _Float16* Ash, int row, int k0, int half) {
  const _Float16* p0 = Ash + row * ASTRIDE + k0 + half * 8;
  const v8h lo = *(const v8h*)p0;
  const v8h hi = *(const v8h*)(p0 + 16);
  v16h a;
#pragma unroll
  for (int e = 0; e < 8; ++e) { a[e] = lo[e]; a[e + 8] = hi[e]; }
  return a;
}

// ---------------------------------------------------------------------------
// Kernel 2: one workgroup per (b,i). 8 waves.
//   t[f,j] = sum_d (Wdep_w[f,d]*q_i[d]) * k_j[d]      -- WMMA f16 GEMM 64x512x256
//   logits[j] = sum_f mask[b,i,j,f]*t[f,j] + sum_d Wdep_b[d]*q_i[d]*k_j[d]
//   out[b,i,:] = softmax(logits)
// ---------------------------------------------------------------------------
__global__ __launch_bounds__(256) void dep_attn_kernel(
    const float* __restrict__ mask,
    const float* __restrict__ Wdep_w, const float* __restrict__ Wdep_b,
    const float* __restrict__ qf, const float* __restrict__ kf,
    const _Float16* __restrict__ kh, float* __restrict__ out) {
  __shared__ _Float16 Ash[Fc * ASTRIDE];   // 33,792 B : A = Wdep_w * diag(q_i), f16
  __shared__ float qrow[HDc];
  __shared__ float qb_sh[HDc];             // Wdep_b .* q_i
  __shared__ float lrow[Nc];               // logits row
  __shared__ float red[256];               // reduction scratch

  const int bi   = blockIdx.x;             // b*N + i
  const int b    = bi >> 9;
  const int t    = threadIdx.x;
  const int w    = t >> 5;                 // wave id 0..7
  const int lane = t & 31;
  const int lan16 = lane & 15;
  const int half  = lane >> 4;

  // q row + bias-weighted q
  {
    const float qv = qf[bi * HDc + t];
    qrow[t]  = qv;
    qb_sh[t] = Wdep_b[t] * qv;
  }
  __syncthreads();

  // Build A[f][d] = Wdep_w[f][d] * q_i[d] in f16 (4 threads per row of 256)
  {
    const int f     = t >> 2;
    const int dbase = (t & 3) * 64;
    const float* wr = Wdep_w + f * HDc + dbase;
    _Float16* ar    = Ash + f * ASTRIDE + dbase;
#pragma unroll 8
    for (int d = 0; d < 64; ++d) ar[d] = (_Float16)(wr[d] * qrow[dbase + d]);
  }
  __syncthreads();

  // Each wave owns 4 consecutive N-tiles of 16 columns: j in [64w, 64w+64)
  for (int nt = 0; nt < 4; ++nt) {
    const int j0 = (w * 4 + nt) * 16;
    const int j  = j0 + lan16;

    // B operand (32x16 f16 layout): lane half h covers K = 8h + e, 16 + 8h + e
    const _Float16* kb  = kh + (size_t)(b * Nc + j) * HDc + half * 8;
    const float*   mrow = mask + ((size_t)bi * Nc + j) * Fc;  // 64 floats
    __builtin_prefetch(mrow, 0, 3);       // near-cache prefetch, overlaps WMMA loop
    __builtin_prefetch(mrow + 32, 0, 3);

    v8f acc0 = {}, acc1 = {}, acc2 = {}, acc3 = {};   // M-tiles 0..3 (f = 0..63)
    // unroll capped at 2: keeps A fragments transient so VGPR count stays
    // < 256 (no s_set_vgpr_msb churn, higher occupancy for the mask stream)
#pragma unroll 2
    for (int k0 = 0; k0 < HDc; k0 += 32) {
      const v8h blo = *(const v8h*)(kb + k0);
      const v8h bhi = *(const v8h*)(kb + k0 + 16);
      const v16h bf = __builtin_shufflevector(blo, bhi, 0, 1, 2, 3, 4, 5, 6, 7, 8, 9, 10, 11, 12, 13, 14, 15);
      const v16h a0 = load_afrag(Ash,  0 + lan16, k0, half);
      const v16h a1 = load_afrag(Ash, 16 + lan16, k0, half);
      const v16h a2 = load_afrag(Ash, 32 + lan16, k0, half);
      const v16h a3 = load_afrag(Ash, 48 + lan16, k0, half);
      acc0 = __builtin_amdgcn_wmma_f32_16x16x32_f16(false, a0, false, bf, (short)0, acc0, false, false);
      acc1 = __builtin_amdgcn_wmma_f32_16x16x32_f16(false, a1, false, bf, (short)0, acc1, false, false);
      acc2 = __builtin_amdgcn_wmma_f32_16x16x32_f16(false, a2, false, bf, (short)0, acc2, false, false);
      acc3 = __builtin_amdgcn_wmma_f32_16x16x32_f16(false, a3, false, bf, (short)0, acc3, false, false);
    }

    // Mask contraction. C/D layout: lane(half,n) VGPR r of tile mt holds
    //   t[f = 16*mt + 8*half + r][j = j0 + lan16]
    float partial = 0.f;
    {
      const float* mp = mrow + half * 8;
      const v8f m0 = *(const v8f*)(mp);
      const v8f m1 = *(const v8f*)(mp + 16);
      const v8f m2 = *(const v8f*)(mp + 32);
      const v8f m3 = *(const v8f*)(mp + 48);
#pragma unroll
      for (int r = 0; r < 8; ++r)
        partial += m0[r] * acc0[r] + m1[r] * acc1[r] + m2[r] * acc2[r] + m3[r] * acc3[r];
    }
    // Bias term (Wdep_b .* q_i) . k_j, d-range split across half-waves
    {
      const v4f* kp   = (const v4f*)(kf + (size_t)(b * Nc + j) * HDc + half * 128);
      const float* qp = qb_sh + half * 128;
#pragma unroll 4
      for (int c = 0; c < 32; ++c) {
        const v4f kv = kp[c];
        partial += qp[c*4+0]*kv[0] + qp[c*4+1]*kv[1] + qp[c*4+2]*kv[2] + qp[c*4+3]*kv[3];
      }
    }
    const float tot = partial + __shfl_xor(partial, 16, 32);
    if (half == 0) lrow[j] = tot;
  }
  __syncthreads();

  // Softmax over lrow[0..511]
  const float l0 = lrow[t];
  const float l1 = lrow[t + 256];
  red[t] = fmaxf(l0, l1);
  __syncthreads();
#pragma unroll
  for (int s = 128; s > 0; s >>= 1) {
    if (t < s) red[t] = fmaxf(red[t], red[t + s]);
    __syncthreads();
  }
  const float mx = red[0];
  __syncthreads();
  const float e0 = __expf(l0 - mx);
  const float e1 = __expf(l1 - mx);
  red[t] = e0 + e1;
  __syncthreads();
#pragma unroll
  for (int s = 128; s > 0; s >>= 1) {
    if (t < s) red[t] += red[t + s];
    __syncthreads();
  }
  const float inv = 1.0f / red[0];
  float* orow = out + (size_t)bi * Nc;
  orow[t]       = e0 * inv;
  orow[t + 256] = e1 * inv;
}

// ---------------------------------------------------------------------------
extern "C" void kernel_launch(void* const* d_in, const int* in_sizes, int n_in,
                              void* d_out, int out_size, void* d_ws, size_t ws_size,
                              hipStream_t stream) {
  const float* x      = (const float*)d_in[0];
  const float* mask   = (const float*)d_in[1];
  const float* Wq_w   = (const float*)d_in[2];
  const float* Wq_b   = (const float*)d_in[3];
  const float* Wk_w   = (const float*)d_in[4];
  const float* Wk_b   = (const float*)d_in[5];
  // d_in[6], d_in[7] = Wv_w, Wv_b : dead code in the reference, skipped
  const float* Wdep_w = (const float*)d_in[8];
  const float* Wdep_b = (const float*)d_in[9];
  float* out = (float*)d_out;

  // workspace: q f32 | k f32 | k f16   (1MB + 1MB + 0.5MB)
  float*    qf = (float*)d_ws;
  float*    kf = qf + Bc * Nc * HDc;
  _Float16* kh = (_Float16*)(kf + Bc * Nc * HDc);

  proj_kernel<<<Bc * Nc, 256, 0, stream>>>(x, Wq_w, Wq_b, Wk_w, Wk_b, qf, kf, kh);
  dep_attn_kernel<<<Bc * Nc, 256, 0, stream>>>(mask, Wdep_w, Wdep_b, qf, kf, kh, out);
}
